// ValueNetwork_68453188764146
// MI455X (gfx1250) — compile-verified, matrix operand order checked by reading, not run on hardware
//
#include <hip/hip_runtime.h>

typedef __attribute__((ext_vector_type(16))) _Float16 v16h;
typedef __attribute__((ext_vector_type(8)))  _Float16 v8h;
typedef __attribute__((ext_vector_type(8)))  float    v8f;

#define NEGV (-9.0e15f)

// Load a 16x32 f16 WMMA A/B fragment from LDS (row-major, stride_h halfs).
// Per CDNA5 ISA 16-bit A layout: lane l -> row (l&15); K runs
// [kb, kb+8) and [16+kb, 16+kb+8), kb = 8*(l>>4). Two ds_load_b128 each.
__device__ __forceinline__ v16h load_frag16(const _Float16* base, int stride_h,
                                            int row0, int koff, int lane) {
  int r  = row0 + (lane & 15);
  int kb = (lane >> 4) << 3;
  const _Float16* p = base + r * stride_h + koff + kb;
  v8h lo = *(const v8h*)(p);
  v8h hi = *(const v8h*)(p + 16);
  return __builtin_shufflevector(lo, hi, 0,1,2,3,4,5,6,7,8,9,10,11,12,13,14,15);
}

__global__ __launch_bounds__(256, 1)
void vnet_gat_kernel(const float* __restrict__ state,
                     const float* __restrict__ wr_w0, const float* __restrict__ wr_b0,
                     const float* __restrict__ wr_w1, const float* __restrict__ wr_b1,
                     const float* __restrict__ wh_w0, const float* __restrict__ wh_b0,
                     const float* __restrict__ wh_w1, const float* __restrict__ wh_b1,
                     const float* __restrict__ g0_w0, const float* __restrict__ g0_b0,
                     const float* __restrict__ g0_w1, const float* __restrict__ g0_b1,
                     const float* __restrict__ g1_w0, const float* __restrict__ g1_b0,
                     const float* __restrict__ g1_w1, const float* __restrict__ g1_b1,
                     const float* __restrict__ v_w0, const float* __restrict__ v_b0,
                     const float* __restrict__ v_w1, const float* __restrict__ v_b1,
                     const float* __restrict__ v_w2, const float* __restrict__ v_b2,
                     float* __restrict__ out) {
  // ---- LDS staging (~86 KB; ~3 concurrent WGs per 320KB WGP) ----
  __shared__ __align__(16) _Float16 Xh[64][32];    // X (rows padded to 64) f16, A-side
  __shared__ __align__(16) _Float16 XhT[32][64];   // X^T f16, B-side for att@X
  __shared__ __align__(16) _Float16 WaT[64][32];   // W0_top^T  (n,k) f16
  __shared__ __align__(16) _Float16 WbT[64][32];   // W0_bot^T  (n,k) f16
  __shared__ __align__(16) float    U[64][64];     // X@W0_top + b0 (also emb hidden)
  __shared__ __align__(16) float    V[64][64];     // X@W0_bot
  __shared__ __align__(16) float    Att[51][64];   // attention logits / exp scratch
  __shared__ __align__(16) _Float16 AttH[64][64];  // normalized attention f16 (padded)
  __shared__ __align__(16) float    Hout[64][32];  // GAT output
  __shared__ __align__(16) float    Res[64][32];   // running X + H1 + H2
  __shared__ float w1s[64], b0s[64];
  __shared__ float b1sh;
  __shared__ float t1[152], t2[104];

  const int tid  = threadIdx.x;
  const int lane = tid & 31;
  const int wid  = tid >> 5;
  const int b    = blockIdx.x;
  const float* st = state + (size_t)b * 750;   // (50,15) row-major

  __builtin_prefetch(st, 0, 1);                // global_prefetch_b8

  // Zero AttH pads once (valid region rewritten each GAT layer).
  for (int i = tid; i < 64 * 64; i += 256) ((_Float16*)AttH)[i] = (_Float16)0.0f;

  // ---- Embeddings, layer 1 -> U used as hidden (51 x 64) ----
  for (int i = tid; i < 51 * 64; i += 256) {
    int row = i >> 6, j = i & 63;
    float acc;
    if (row == 0) {
      acc = wr_b0[j];
      #pragma unroll
      for (int k = 0; k < 9; ++k) acc += st[k] * wr_w0[k * 64 + j];
    } else {
      acc = wh_b0[j];
      const float* hp = st + (row - 1) * 15 + 9;
      #pragma unroll
      for (int k = 0; k < 6; ++k) acc += hp[k] * wh_w0[k * 64 + j];
    }
    U[row][j] = fmaxf(acc, 0.0f);
  }
  __syncthreads();

  // ---- Embeddings, layer 2 -> Res (=X), Xh, XhT; rows 51..63 zeroed ----
  for (int i = tid; i < 64 * 32; i += 256) {
    int row = i >> 5, f = i & 31;
    float val = 0.0f;
    if (row < 51) {
      const float* W = (row == 0) ? wr_w1 : wh_w1;
      float acc = (row == 0) ? wr_b1[f] : wh_b1[f];
      for (int k = 0; k < 64; ++k) acc += U[row][k] * W[k * 32 + f];
      val = fmaxf(acc, 0.0f);
    }
    Res[row][f] = val;
    _Float16 h = (_Float16)val;
    Xh[row][f] = h;
    XhT[f][row] = h;
  }
  __syncthreads();

  #pragma unroll
  for (int layer = 0; layer < 2; ++layer) {
    const float* gw0 = (layer == 0) ? g0_w0 : g1_w0;
    const float* gb0 = (layer == 0) ? g0_b0 : g1_b0;
    const float* gw1 = (layer == 0) ? g0_w1 : g1_w1;
    const float* gb1 = (layer == 0) ? g0_b1 : g1_b1;

    // ---- Stage GAT weights (split + transpose to f16) ----
    for (int i = tid; i < 64 * 64; i += 256) {
      int k = i >> 6, n = i & 63;
      _Float16 w = (_Float16)gw0[i];
      if (k < 32) WaT[n][k] = w; else WbT[n][k - 32] = w;
    }
    if (tid < 64) { w1s[tid] = gw1[tid]; b0s[tid] = gb0[tid]; }
    if (tid == 0) b1sh = gb1[0];
    __syncthreads();

    // ---- U = X @ W0_top + b0,  V = X @ W0_bot  via WMMA (64x64, K=32) ----
    // 16 output tiles, 8 waves -> statically unrolled 2 tiles per wave
    // (keeps trip count compile-time so no exec-mask loop machinery).
    #pragma unroll
    for (int tt = 0; tt < 2; ++tt) {
      int t  = wid + (tt << 3);
      int mt = t >> 2, nt = t & 3;
      v16h a  = load_frag16(&Xh[0][0],  32, mt * 16, 0, lane);
      v16h ba = load_frag16(&WaT[0][0], 32, nt * 16, 0, lane);
      v16h bb = load_frag16(&WbT[0][0], 32, nt * 16, 0, lane);
      v8f cu = {}, cv = {};
      cu = __builtin_amdgcn_wmma_f32_16x16x32_f16(false, a, false, ba, (short)0, cu, false, false);
      cv = __builtin_amdgcn_wmma_f32_16x16x32_f16(false, a, false, bb, (short)0, cv, false, false);
      int n  = nt * 16 + (lane & 15);
      int m0 = mt * 16 + ((lane >> 4) << 3);
      float bias = b0s[n];
      #pragma unroll
      for (int v = 0; v < 8; ++v) {
        U[m0 + v][n] = cu[v] + bias;
        V[m0 + v][n] = cv[v];
      }
    }
    __syncthreads();

    // ---- Pairwise logits: A_ij = b1 + sum_k relu(U_ik + V_jk) * w1_k ----
    float bb1 = b1sh;
    for (int i = tid; i < 51 * 51; i += 256) {
      int ii = i / 51, jj = i - ii * 51;
      float acc = bb1;
      for (int k = 0; k < 64; ++k) {
        float s = U[ii][k] + V[jj][k];
        acc += fmaxf(s, 0.0f) * w1s[k];
      }
      float e = (acc > 0.0f) ? acc : 0.04f * acc;   // leaky-relu
      if (jj == 0 && ii > 0) e = NEGV;              // adj mask: adj[1:,0]=0
      Att[ii][jj] = e;
    }
    __syncthreads();

    // ---- Row softmax -> AttH (f16, zero-padded to 64x64) ----
    if (tid < 51) {
      float m = -3.4e38f;
      for (int j = 0; j < 51; ++j) m = fmaxf(m, Att[tid][j]);
      float s = 0.0f;
      for (int j = 0; j < 51; ++j) { float e = __expf(Att[tid][j] - m); Att[tid][j] = e; s += e; }
      float inv = 1.0f / s;
      for (int j = 0; j < 51; ++j) AttH[tid][j] = (_Float16)(Att[tid][j] * inv);
    }
    __syncthreads();

    // ---- H = att @ X via WMMA (64x32, K=64 = 2 k-steps), 8 tiles / 8 waves ----
    {
      int mt = wid >> 1, nt = wid & 1;
      v8f c = {};
      #pragma unroll
      for (int ks = 0; ks < 2; ++ks) {
        v16h a  = load_frag16(&AttH[0][0], 64, mt * 16, ks * 32, lane);
        v16h bx = load_frag16(&XhT[0][0],  64, nt * 16, ks * 32, lane);
        c = __builtin_amdgcn_wmma_f32_16x16x32_f16(false, a, false, bx, (short)0, c, false, false);
      }
      int n  = nt * 16 + (lane & 15);
      int m0 = mt * 16 + ((lane >> 4) << 3);
      #pragma unroll
      for (int v = 0; v < 8; ++v) {
        Hout[m0 + v][n] = c[v];
        Res[m0 + v][n] += c[v];                     // residual accumulation
      }
    }
    __syncthreads();

    // ---- Next layer input: X <- H1 ----
    if (layer == 0) {
      for (int i = tid; i < 51 * 32; i += 256) {
        int r = i >> 5, f = i & 31;
        _Float16 h = (_Float16)Hout[r][f];
        Xh[r][f] = h;
        XhT[f][r] = h;
      }
      __syncthreads();
    }
  }

  // ---- Value head on node 0: 32 -> 150 -> 100 -> 1 (all ReLU) ----
  if (tid < 150) {
    float acc = v_b0[tid];
    for (int k = 0; k < 32; ++k) acc += Res[0][k] * v_w0[k * 150 + tid];
    t1[tid] = fmaxf(acc, 0.0f);
  }
  __syncthreads();
  if (tid < 100) {
    float acc = v_b1[tid];
    for (int k = 0; k < 150; ++k) acc += t1[k] * v_w1[k * 100 + tid];
    t2[tid] = fmaxf(acc, 0.0f);
  }
  __syncthreads();
  if (tid == 0) {
    float acc = v_b2[0];
    for (int k = 0; k < 100; ++k) acc += t2[k] * v_w2[k];
    out[b] = fmaxf(acc, 0.0f);
  }
}

extern "C" void kernel_launch(void* const* d_in, const int* in_sizes, int n_in,
                              void* d_out, int out_size, void* d_ws, size_t ws_size,
                              hipStream_t stream) {
  const float* state = (const float*)d_in[0];
  // d_in[1] = dropout (unused, == 0)
  const float* wr_w0 = (const float*)d_in[2];
  const float* wr_b0 = (const float*)d_in[3];
  const float* wr_w1 = (const float*)d_in[4];
  const float* wr_b1 = (const float*)d_in[5];
  const float* wh_w0 = (const float*)d_in[6];
  const float* wh_b0 = (const float*)d_in[7];
  const float* wh_w1 = (const float*)d_in[8];
  const float* wh_b1 = (const float*)d_in[9];
  const float* g0_w0 = (const float*)d_in[10];
  const float* g0_b0 = (const float*)d_in[11];
  const float* g0_w1 = (const float*)d_in[12];
  const float* g0_b1 = (const float*)d_in[13];
  const float* g1_w0 = (const float*)d_in[14];
  const float* g1_b0 = (const float*)d_in[15];
  const float* g1_w1 = (const float*)d_in[16];
  const float* g1_b1 = (const float*)d_in[17];
  const float* v_w0  = (const float*)d_in[18];
  const float* v_b0  = (const float*)d_in[19];
  const float* v_w1  = (const float*)d_in[20];
  const float* v_b1  = (const float*)d_in[21];
  const float* v_w2  = (const float*)d_in[22];
  const float* v_b2  = (const float*)d_in[23];

  const int B = in_sizes[0] / (50 * 15);   // 512

  vnet_gat_kernel<<<B, 256, 0, stream>>>(
      state,
      wr_w0, wr_b0, wr_w1, wr_b1,
      wh_w0, wh_b0, wh_w1, wh_b1,
      g0_w0, g0_b0, g0_w1, g0_b1,
      g1_w0, g1_b0, g1_w1, g1_b1,
      v_w0, v_b0, v_w1, v_b1, v_w2, v_b2,
      (float*)d_out);
}
